// FastGRNNCUDA_29222957482091
// MI455X (gfx1250) — matrix operand order repaired and checked by reading, not compiled, not run on hardware
//
#include <hip/hip_runtime.h>
#include <hip/hip_bf16.h>

// ---------------------------------------------------------------------------
// FastGRNN on MI455X (gfx1250).
//   Phase 1: t1[T*B, RW] = x @ W1^T               (bf16 WMMA, f32 accum)
//   Phase 2: persistent scan kernel (16 blocks x 16 waves), per step t:
//       a   = h @ U1^T         stage A: 64 wave-tiles, K=1024, U1 from LDS
//       pre = t1_t @ W2^T + a @ U2^T   stage B: 4 tiles/wave, W2/U2 from LDS
//       z = sigmoid(pre+bg); c = tanh(pre+bu)
//       h  = (zs*(1-z)+ns)*c + z*h  (h lives in REGISTERS across all steps)
//   2 device-wide barriers per step (sense-reversal, 16 arrivals).
//   Weight staging uses GLOBAL_LOAD_ASYNC_TO_LDS_B128 when available.
// ---------------------------------------------------------------------------

typedef __attribute__((ext_vector_type(16))) __bf16 v16bf;
typedef __attribute__((ext_vector_type(8)))  __bf16 v8bf;
typedef __attribute__((ext_vector_type(8)))  float  v8f;

#define T_STEPS 512
#define BATCH   256
#define DIM     256
#define HID     1024
#define RW_     64
#define RU_     64
#define NBLOCKS 16        // persistent blocks (256 waves), trivially co-resident
#define SCAN_THREADS 512  // 16 waves per block

// workspace layout (bytes, 256-aligned)
static constexpr size_t OFF_BAR = 0;                      // 2x u32 barrier state
static constexpr size_t OFF_PW1 = 256;                    // D*RW bf16   = 32768 B
static constexpr size_t OFF_PW2 = OFF_PW1 + 32768;        // RW*H bf16   = 131072 B
static constexpr size_t OFF_PU1 = OFF_PW2 + 131072;       // H*RU bf16   = 131072 B
static constexpr size_t OFF_PU2 = OFF_PU1 + 131072;       // RU*H bf16   = 131072 B
static constexpr size_t OFF_T1  = OFF_PU2 + 131072;       // T*B*RW bf16 = 16 MB
static constexpr size_t OFF_HBF = OFF_T1  + (size_t)T_STEPS*BATCH*RW_*2;  // B*H bf16
static constexpr size_t OFF_ABF = OFF_HBF + (size_t)BATCH*HID*2;          // B*RU bf16

// LDS layout inside scan kernel (dynamic shared)
static constexpr unsigned LDS_U1  = 0;                     // full pU1: 131072 B
static constexpr unsigned LDS_W2S = 131072;                // W2 slice:   8192 B
static constexpr unsigned LDS_U2S = 131072 + 8192;         // U2 slice:   8192 B
static constexpr unsigned LDS_TOT = 131072 + 8192 + 8192;  // 147456 B < 320 KB

// ---------------------------------------------------------------------------
// Async global -> LDS staging (CDNA5 ASYNCcnt path) with safe fallback.
// Builtin signature (from hipcc diagnostic): arg0 is a non-const int4-vector
// pointer in address space 1 (global); arg1 the LDS-side pointer; then
// immediate offset and cpol.
// ---------------------------------------------------------------------------
#if __has_builtin(__builtin_amdgcn_global_load_async_to_lds_b128) && \
    __has_builtin(__builtin_amdgcn_s_wait_asynccnt)
#define USE_ASYNC_LDS 1
typedef int v4i_ __attribute__((vector_size(16)));
typedef __attribute__((address_space(1))) v4i_* as1_v4i_ptr;
typedef __attribute__((address_space(3))) v4i_* as3_v4i_ptr;
#else
#define USE_ASYNC_LDS 0
#endif

__device__ __forceinline__ void stage_copy(const void* __restrict__ g,
                                           void* __restrict__ s, int bytes) {
#if USE_ASYNC_LDS
  for (int idx = threadIdx.x; idx < bytes / 16; idx += (int)blockDim.x) {
    __builtin_amdgcn_global_load_async_to_lds_b128(
        (as1_v4i_ptr)((const char*)g + idx * 16),
        (as3_v4i_ptr)((char*)s + idx * 16), 0, 0);
  }
#else
  const uint4* gs = (const uint4*)g;
  uint4* ss = (uint4*)s;
  for (int idx = threadIdx.x; idx < bytes / 16; idx += (int)blockDim.x)
    ss[idx] = gs[idx];
#endif
}

__device__ __forceinline__ void stage_copy_wait() {
#if USE_ASYNC_LDS
  __builtin_amdgcn_s_wait_asynccnt(0);
#endif
  __syncthreads();
}

// ---------------------------------------------------------------------------
// WMMA helpers (layouts per CDNA5 ISA 7.12.2)
// ---------------------------------------------------------------------------
__device__ __forceinline__ v8f wmma_bf16(v16bf a, v16bf b, v8f c) {
  return __builtin_amdgcn_wmma_f32_16x16x32_bf16(false, a, false, b, (short)0, c,
                                                 false, false);
}

// A-operand fragment from row-major bf16 [rows x ld]; lane half selects K
// sub-blocks: elements 0..7 = K(k0 + half*8 + i), 8..15 = K(k0 + 16 + half*8 + i)
__device__ __forceinline__ v16bf load_a_frag(const __bf16* __restrict__ base,
                                             int ld, int row, int k0, int half) {
  const __bf16* rp = base + (size_t)row * ld + k0 + half * 8;
  v8bf lo = *(const v8bf*)(rp);
  v8bf hi = *(const v8bf*)(rp + 16);
  v16bf r;
#pragma unroll
  for (int i = 0; i < 8; ++i) { r[i] = lo[i]; r[8 + i] = hi[i]; }
  return r;
}

// B-operand fragment: contiguous 16 bf16 per lane, ((kt*ntiles + nt)*32 + lane)*16
__device__ __forceinline__ v16bf load_b_frag(const __bf16* pk, int ntiles,
                                             int kt, int nt, int lane) {
  return *(const v16bf*)(pk + (((size_t)kt * ntiles + nt) * 32 + lane) * 16);
}

// ---------------------------------------------------------------------------
// Weight packing: Y = A @ Wsrc^T with Wsrc [N,K] row-major f32.
// Packed B(k,n) = Wsrc[n*K+k], k = kt*32 + (lane/16)*8 + (e%8) + (e/8)*16, n = nt*16 + lane%16.
// ---------------------------------------------------------------------------
__global__ void pack_b_kernel(const float* __restrict__ src, __bf16* __restrict__ dst,
                              int K, int N) {
  int idx = blockIdx.x * blockDim.x + threadIdx.x;
  if (idx >= K * N) return;
  int e      = idx & 15;
  int lane   = (idx >> 4) & 31;
  int rest   = idx >> 9;
  int ntiles = N >> 4;
  int nt     = rest % ntiles;
  int kt     = rest / ntiles;
  int k = kt * 32 + (lane >> 4) * 8 + (e & 7) + (e >> 3) * 16;
  int n = nt * 16 + (lane & 15);
  dst[idx] = (__bf16)src[(size_t)n * K + k];
}

__global__ void init_state_kernel(const float* __restrict__ h0,
                                  __bf16* __restrict__ hbf,
                                  unsigned* __restrict__ bar) {
  int idx = blockIdx.x * blockDim.x + threadIdx.x;
  if (idx == 0) bar[0] = 0u;   // arrival counter must start at 0 (ws is poisoned)
  if (idx < BATCH * HID) hbf[idx] = (__bf16)h0[idx];
}

// ---------------------------------------------------------------------------
// Phase 1: t1[T*B, RW] = x[T*B, D] @ W1^T   (one 16x16 tile per wave)
// ---------------------------------------------------------------------------
__global__ void gemm_xw1_kernel(const float* __restrict__ x,
                                const __bf16* __restrict__ pW1,
                                __bf16* __restrict__ t1) {
  int wave = (blockIdx.x * blockDim.x + threadIdx.x) >> 5;
  int lane = threadIdx.x & 31;
  int half = lane >> 4;
  int lrow = lane & 15;
  int mt = wave >> 2;   // (T*B)/16 = 8192 tiles
  int nt = wave & 3;    // RW/16    = 4 tiles
  int row = mt * 16 + lrow;

  v8f acc = {};
#pragma unroll
  for (int kt = 0; kt < DIM / 32; ++kt) {
    const float* rp = x + (size_t)row * DIM + kt * 32 + half * 8;
    v8f lo = *(const v8f*)(rp);
    v8f hi = *(const v8f*)(rp + 16);
    v16bf a;
#pragma unroll
    for (int i = 0; i < 8; ++i) { a[i] = (__bf16)lo[i]; a[8 + i] = (__bf16)hi[i]; }
    v16bf b = load_b_frag(pW1, RW_ / 16, kt, nt, lane);
    acc = wmma_bf16(a, b, acc);
  }
  int col = nt * 16 + lrow;
#pragma unroll
  for (int v = 0; v < 8; ++v) {
    int r = mt * 16 + v + 8 * half;
    t1[(size_t)r * RW_ + col] = (__bf16)acc[v];
  }
}

// ---------------------------------------------------------------------------
// Device-wide barrier (sense via monotonically increasing generation)
// ---------------------------------------------------------------------------
__device__ __forceinline__ void grid_barrier(unsigned* __restrict__ bar) {
  __syncthreads();
  __threadfence();
  if (threadIdx.x == 0) {
    unsigned g = __atomic_load_n(bar + 1, __ATOMIC_RELAXED);
    unsigned a = atomicAdd(bar, 1u);
    if (a == NBLOCKS - 1) {
      bar[0] = 0u;
      __threadfence();
      __atomic_store_n(bar + 1, g + 1u, __ATOMIC_RELAXED);
    } else {
      while (__atomic_load_n(bar + 1, __ATOMIC_RELAXED) == g) {
        __builtin_amdgcn_s_sleep(1);
      }
    }
  }
  __threadfence();
  __syncthreads();
}

// ---------------------------------------------------------------------------
// Phase 2: persistent recurrent scan.
//   Tile map: block owns th in [4*blk, 4*blk+3]; wave owns tb = wave (0..15).
//   Per-block LDS: full U1 (128 KB) + 8 KB slices of W2/U2.
// ---------------------------------------------------------------------------
__global__ void __launch_bounds__(SCAN_THREADS)
scan_kernel(const float* __restrict__ h0,
            const __bf16* __restrict__ t1, const __bf16* __restrict__ pW2,
            const __bf16* __restrict__ pU1, const __bf16* __restrict__ pU2,
            const float* __restrict__ bias_gate, const float* __restrict__ bias_update,
            const float* __restrict__ zeta, const float* __restrict__ nu,
            __bf16* __restrict__ hbf, __bf16* __restrict__ abf,
            float* __restrict__ out, unsigned* __restrict__ bar) {
  extern __shared__ char smem[];
  __bf16* sU1  = (__bf16*)(smem + LDS_U1);
  __bf16* sW2s = (__bf16*)(smem + LDS_W2S);
  __bf16* sU2s = (__bf16*)(smem + LDS_U2S);

  int wave  = threadIdx.x >> 5;                    // 0..15  (== tb)
  int gwave = blockIdx.x * 16 + wave;              // 0..255
  int lane  = threadIdx.x & 31;
  int half  = lane >> 4;
  int lrow  = lane & 15;

  // ---- stage LDS: full pU1 + this block's th-slice of pW2/pU2 (async) ----
  stage_copy(pU1, sU1, HID * RU_ * 2);
#pragma unroll
  for (int kt = 0; kt < 2; ++kt) {
    size_t goff = ((size_t)kt * 64 + blockIdx.x * 4) * 32 * 16;
    stage_copy(pW2 + goff, sW2s + (size_t)kt * 4 * 32 * 16, 4 * 32 * 16 * 2);
    stage_copy(pU2 + goff, sU2s + (size_t)kt * 4 * 32 * 16, 4 * 32 * 16 * 2);
  }
  stage_copy_wait();

  float zs = 1.0f / (1.0f + __expf(-zeta[0]));
  float ns = 1.0f / (1.0f + __expf(-nu[0]));

  // ---- hoist biases and keep h in registers across all timesteps ----
  float bgv[4], buv[4];
#pragma unroll
  for (int i = 0; i < 4; ++i) {
    int col = (blockIdx.x * 4 + i) * 16 + lrow;
    bgv[i] = bias_gate[col];
    buv[i] = bias_update[col];
  }
  float hreg[4][8];
#pragma unroll
  for (int i = 0; i < 4; ++i)
#pragma unroll
    for (int v = 0; v < 8; ++v) {
      int r   = wave * 16 + v + 8 * half;
      int col = (blockIdx.x * 4 + i) * 16 + lrow;
      hreg[i][v] = h0[(size_t)r * HID + col];
    }

  for (int t = 0; t < T_STEPS; ++t) {
    // prefetch next step's t1 rows (global_prefetch_b8)
    if (t + 1 < T_STEPS)
      __builtin_prefetch(t1 + (size_t)(t + 1) * BATCH * RW_ +
                         (size_t)(wave * 16 + lrow) * RW_, 0, 1);

    // ---- stage A: a[B,RU] = h @ U1^T (64 wave-tiles, U1 from LDS) ----
    if ((gwave & 3) == 0) {            // wave-uniform branch, EXEC stays all-1
      int wid = gwave >> 2;            // 0..63
      int tbA = wid >> 2;              // 16 B-tiles
      int tr  = wid & 3;               // 4 RU-tiles
      int rowA = tbA * 16 + lrow;
      v8f acc = {};
#pragma unroll 8
      for (int kt = 0; kt < HID / 32; ++kt) {
        v16bf a = load_a_frag(hbf, HID, rowA, kt * 32, half);
        v16bf b = load_b_frag(sU1, RU_ / 16, kt, tr, lane);
        acc = wmma_bf16(a, b, acc);
      }
      int colA = tr * 16 + lrow;
#pragma unroll
      for (int v = 0; v < 8; ++v)
        abf[(tbA * 16 + v + 8 * half) * RU_ + colA] = (__bf16)acc[v];
    }
    grid_barrier(bar);

    // ---- stage B: pre = t1_t @ W2^T + a @ U2^T, gates, h update ----
    const __bf16* t1t = t1 + (size_t)t * BATCH * RW_;
    int row = wave * 16 + lrow;
    v16bf a_t1_0 = load_a_frag(t1t, RW_, row, 0,  half);
    v16bf a_t1_1 = load_a_frag(t1t, RW_, row, 32, half);
    v16bf a_ab_0 = load_a_frag(abf, RU_, row, 0,  half);
    v16bf a_ab_1 = load_a_frag(abf, RU_, row, 32, half);
#pragma unroll
    for (int i = 0; i < 4; ++i) {
      v8f acc = {};
      acc = wmma_bf16(a_t1_0, load_b_frag(sW2s, 4, 0, i, lane), acc);
      acc = wmma_bf16(a_t1_1, load_b_frag(sW2s, 4, 1, i, lane), acc);
      acc = wmma_bf16(a_ab_0, load_b_frag(sU2s, 4, 0, i, lane), acc);
      acc = wmma_bf16(a_ab_1, load_b_frag(sU2s, 4, 1, i, lane), acc);
      int col = (blockIdx.x * 4 + i) * 16 + lrow;
#pragma unroll
      for (int v = 0; v < 8; ++v) {
        int r     = wave * 16 + v + 8 * half;
        float pre = acc[v];
        float z   = 1.0f / (1.0f + __expf(-(pre + bgv[i])));
        float c   = tanhf(pre + buv[i]);
        float hn  = (zs * (1.0f - z) + ns) * c + z * hreg[i][v];
        hreg[i][v] = hn;
        hbf[r * HID + col] = (__bf16)hn;
        __builtin_nontemporal_store(hn, out + ((size_t)t * BATCH + r) * HID + col);
      }
    }
    grid_barrier(bar);
  }
}

// ---------------------------------------------------------------------------
extern "C" void kernel_launch(void* const* d_in, const int* in_sizes, int n_in,
                              void* d_out, int out_size, void* d_ws, size_t ws_size,
                              hipStream_t stream) {
  (void)in_sizes; (void)n_in; (void)out_size; (void)ws_size;
  const float* x    = (const float*)d_in[0];
  const float* h0   = (const float*)d_in[1];
  const float* W1   = (const float*)d_in[2];
  const float* W2   = (const float*)d_in[3];
  const float* U1   = (const float*)d_in[4];
  const float* U2   = (const float*)d_in[5];
  const float* bg   = (const float*)d_in[6];
  const float* bu   = (const float*)d_in[7];
  const float* zeta = (const float*)d_in[8];
  const float* nu   = (const float*)d_in[9];
  float* out = (float*)d_out;

  char* ws = (char*)d_ws;
  unsigned* bar = (unsigned*)(ws + OFF_BAR);
  __bf16* pW1 = (__bf16*)(ws + OFF_PW1);
  __bf16* pW2 = (__bf16*)(ws + OFF_PW2);
  __bf16* pU1 = (__bf16*)(ws + OFF_PU1);
  __bf16* pU2 = (__bf16*)(ws + OFF_PU2);
  __bf16* t1  = (__bf16*)(ws + OFF_T1);
  __bf16* hbf = (__bf16*)(ws + OFF_HBF);
  __bf16* abf = (__bf16*)(ws + OFF_ABF);

  // pack weights into WMMA B-operand layout (bf16)
  pack_b_kernel<<<(DIM * RW_ + 255) / 256, 256, 0, stream>>>(W1, pW1, DIM, RW_);
  pack_b_kernel<<<(RW_ * HID + 255) / 256, 256, 0, stream>>>(W2, pW2, RW_, HID);
  pack_b_kernel<<<(HID * RU_ + 255) / 256, 256, 0, stream>>>(U1, pU1, HID, RU_);
  pack_b_kernel<<<(RU_ * HID + 255) / 256, 256, 0, stream>>>(U2, pU2, RU_, HID);

  // init bf16 h state + barrier counter
  init_state_kernel<<<(BATCH * HID + 255) / 256, 256, 0, stream>>>(h0, hbf, bar);

  // phase 1: t1 = x @ W1^T  (32768 wave-tiles, 8 waves/block)
  gemm_xw1_kernel<<<4096, 256, 0, stream>>>(x, pW1, t1);

  // phase 2: persistent scan (16 blocks x 512 threads, 147456 B dynamic LDS)
  scan_kernel<<<NBLOCKS, SCAN_THREADS, LDS_TOT, stream>>>(h0, t1, pW2, pU1, pU2,
                                                          bg, bu, zeta, nu,
                                                          hbf, abf, out, bar);
}
